// BidirectionalPropagation_22016002359628
// MI455X (gfx1250) — compile-verified, matrix operand order checked by reading, not run on hardware
//
#include <hip/hip_runtime.h>

typedef __attribute__((ext_vector_type(16))) _Float16 v16h;
typedef __attribute__((ext_vector_type(8)))  _Float16 v8h;
typedef __attribute__((ext_vector_type(8)))  float    v8f;

#define CCH 128
#define GDEF 16
#define CG   8
#define HH   60
#define WW   108
#define HWP  (HH*WW)      // 6480 = 405*16
#define NPIXT (HWP/16)    // 405 pixel tiles
#define TT   10
#define WVB  4            // waves per block in conv kernel

#if __has_builtin(__builtin_amdgcn_global_load_async_to_lds_b128) && \
    __has_builtin(__builtin_amdgcn_s_wait_asynccnt)
#define HAS_ASYNC 1
#else
#define HAS_ASYNC 0
#endif

#if HAS_ASYNC
typedef int v4i_vs __attribute__((vector_size(16)));
typedef __attribute__((address_space(1))) v4i_vs* gvp4;   // global int4*
typedef __attribute__((address_space(3))) v4i_vs* lvp4;   // LDS int4*
#define GASYNC_B128(dst, src)                                                  \
    __builtin_amdgcn_global_load_async_to_lds_b128((gvp4)(src), (lvp4)(dst), 0, 0)
#endif

// ---------------------------------------------------------------------------
// Implicit-GEMM conv (3x3 or 1x1) via WMMA f32 += f16 x f16, 4-way N-tiling.
// inF : [HW][Cip] f16 activations, HWC, Cip = Ci padded to mult of 32
// wpf : fragment-packed weights: [((kblk*nCoT + coT)*32 + lane)*16 + e] f16
//       K = tap*Cip + ci ; CoP = Co padded to mult of 64 (zero pad tiles)
// out : [Co][HW] f32 planar.  fuse: 0=none, 1=leaky(slope), 2=+resid
// Wave = 16(pixel) x 64(outch) tile (4 accumulators, A reused 4x).
// ---------------------------------------------------------------------------
template <int TAPS>
__global__ __launch_bounds__(128)
void k_conv_wmma(const _Float16* __restrict__ inF, const _Float16* __restrict__ wpf,
                 const float* __restrict__ bias, const float* __restrict__ resid,
                 float* __restrict__ out,
                 int Cip, int Co, int CoP, int fuse, float slope)
{
    const int lane = threadIdx.x & 31;
    const int wave = threadIdx.x >> 5;
    const int nCoT = CoP >> 4;                    // padded 16-wide co tiles
    const int nCoG = CoP >> 6;                    // groups of 4 tiles
    const int task = blockIdx.x * WVB + wave;
    const int pixT = task / nCoG;
    const int coG  = task - pixT * nCoG;
    if (pixT >= NPIXT) return;                    // wave-uniform
    const int coT0 = coG * 4;
    const int m0   = pixT * 16;
    const int mrow = m0 + (lane & 15);            // A row (pixel) for this lane
    const int py   = mrow / WW;
    const int px   = mrow - py * WW;
    const int hs   = (lane >= 16) ? 8 : 0;        // K half select
    const int nKblk = (TAPS * Cip) >> 5;

#if HAS_ASYNC
    __shared__ __align__(32) _Float16 bstage[2][WVB][4][32][16];   // 32 KB
    {   // prologue: stage B block 0 (4 tiles, 8 x b128)
#pragma unroll
        for (int t = 0; t < 4; ++t) {
            const _Float16* src = wpf + ((size_t)(coT0 + t) * 32 + lane) * 16;
            _Float16* dst = &bstage[0][wave][t][lane][0];
            GASYNC_B128(dst, src);
            GASYNC_B128(dst + 8, src + 8);
        }
    }
#endif

    v8f acc[4] = {};
    int tap = 0, c0 = 0;
    for (int kblk = 0; kblk < nKblk; ++kblk) {
#if HAS_ASYNC
        if (kblk + 1 < nKblk) {   // stage next block while current drains
#pragma unroll
            for (int t = 0; t < 4; ++t) {
                const _Float16* src =
                    wpf + (((size_t)(kblk + 1) * nCoT + coT0 + t) * 32 + lane) * 16;
                _Float16* dst = &bstage[(kblk + 1) & 1][wave][t][lane][0];
                GASYNC_B128(dst, src);
                GASYNC_B128(dst + 8, src + 8);
            }
            __builtin_amdgcn_s_wait_asynccnt(8);   // current block's 8 done
        } else {
            __builtin_amdgcn_s_wait_asynccnt(0);
        }
#endif
        // ---- A fragment: two contiguous v8h loads (HWC layout) ----
        int y = py, x = px;
        if (TAPS == 9) { const int ty = tap / 3; y += ty - 1; x += (tap - ty * 3) - 1; }
        v8h lo = {}, hi = {};
        if (((unsigned)y < (unsigned)HH) & ((unsigned)x < (unsigned)WW)) {
            const _Float16* ap = inF + (long)(y * WW + x) * Cip + hs + c0;
            lo = *(const v8h*)(ap);
            hi = *(const v8h*)(ap + 16);
        }
        v16h a = __builtin_shufflevector(lo, hi, 0,1,2,3,4,5,6,7,8,9,10,11,12,13,14,15);
        // ---- 4 B fragments, 4 WMMAs reusing A ----
#pragma unroll
        for (int t = 0; t < 4; ++t) {
#if HAS_ASYNC
            v16h b = *(const v16h*)&bstage[kblk & 1][wave][t][lane][0];
#else
            if (kblk + 1 < nKblk)
                __builtin_prefetch((const void*)(wpf +
                    (((size_t)(kblk + 1) * nCoT + coT0 + t) * 32 + lane) * 16), 0, 1);
            v16h b = *(const v16h*)(wpf +
                (((size_t)kblk * nCoT + coT0 + t) * 32 + lane) * 16);
#endif
            acc[t] = __builtin_amdgcn_wmma_f32_16x16x32_f16(false, a, false, b,
                                                            (short)0, acc[t], false, false);
        }
        c0 += 32;
        if (c0 == Cip) { c0 = 0; ++tap; }
    }

#pragma unroll
    for (int t = 0; t < 4; ++t) {
        const int coT = coT0 + t;
        if (coT * 16 < Co) {                       // skip zero-padded tiles
            const int colB = coT * 16 + (lane & 15);
            const float bv = bias ? bias[colB] : 0.f;
#pragma unroll
            for (int r = 0; r < 8; ++r) {
                const int m = m0 + r + hs;
                float v = acc[t][r] + bv;
                if (fuse == 1)      v = (v > 0.f) ? v : v * slope;
                else if (fuse == 2) v += resid[(size_t)colB * HWP + m];
                out[(size_t)colB * HWP + m] = v;
            }
        }
    }
}

// ------------------------- small helper kernels ----------------------------
__device__ __forceinline__ float bcorner(const float* __restrict__ p, int yi, int xi) {
    if (yi < 0 || yi > HH - 1 || xi < 0 || xi > WW - 1) return 0.f;
    return p[yi * WW + xi];
}
__device__ __forceinline__ float bilin(const float* __restrict__ p, float xs, float ys) {
    const float x0f = floorf(xs), y0f = floorf(ys);
    const int   x0 = (int)x0f,   y0 = (int)y0f;
    const float wx1 = xs - x0f,  wy1 = ys - y0f;
    const float wx0 = 1.f - wx1, wy0 = 1.f - wy1;
    return bcorner(p, y0,     x0    ) * wy0 * wx0
         + bcorner(p, y0,     x0 + 1) * wy0 * wx1
         + bcorner(p, y0 + 1, x0    ) * wy1 * wx0
         + bcorner(p, y0 + 1, x0 + 1) * wy1 * wx1;
}

// zero-fill f16 buffer
__global__ __launch_bounds__(256)
void k_zero16(_Float16* __restrict__ dst, int n) {
    int i = blockIdx.x * blockDim.x + threadIdx.x;
    if (i < n) dst[i] = (_Float16)0.f;
}

// planar f32 [C][HW] -> HWC f16 [HW][Cip] at channel offset (realizes concat)
__global__ __launch_bounds__(256)
void k_f16hwc(const float* __restrict__ src, _Float16* __restrict__ dst,
              int chanOff, int C, int Cip) {
    int i = blockIdx.x * blockDim.x + threadIdx.x;
    if (i >= C * HWP) return;
    const int p = i / C;
    const int c = i - p * C;
    dst[(size_t)p * Cip + chanOff + c] = (_Float16)src[(size_t)c * HWP + p];
}

// OIHW f32 -> fragment-packed f16, K = tap*Cip + ci; zero pad ci>=Ci, o>=Co
__global__ __launch_bounds__(256)
void k_pack_frag(const float* __restrict__ w, _Float16* __restrict__ wpf,
                 int Ci, int Cip, int Co, int CoP) {
    const int nCoT = CoP >> 4;
    const int n = 9 * Cip * CoP;
    int i = blockIdx.x * blockDim.x + threadIdx.x;
    if (i >= n) return;
    const int e = i & 15;
    const int l = (i >> 4) & 31;
    const int r = i >> 9;
    const int coT  = r % nCoT;
    const int kblk = r / nCoT;
    const int K   = kblk * 32 + ((l >= 16) ? 8 : 0) + e + ((e < 8) ? 0 : 8);
    const int tap = K / Cip;
    const int ci  = K - tap * Cip;
    const int o   = coT * 16 + (l & 15);
    const float v = (ci < Ci && o < Co) ? w[((size_t)o * Ci + ci) * 9 + tap] : 0.f;
    wpf[i] = (_Float16)v;
}

// da_w (C,C,3,3) -> fragment-packed, 1x1 GEMM with K = (g*9+k)*8 + c
__global__ __launch_bounds__(256)
void k_pack_daw_frag(const float* __restrict__ w, _Float16* __restrict__ wpf) {
    const int nCoT = 8;                       // Co = CoP = 128
    const int n = 1152 * 128;
    int i = blockIdx.x * blockDim.x + threadIdx.x;
    if (i >= n) return;
    const int e = i & 15;
    const int l = (i >> 4) & 31;
    const int r = i >> 9;
    const int coT  = r % nCoT;
    const int kblk = r / nCoT;
    const int K  = kblk * 32 + ((l >= 16) ? 8 : 0) + e + ((e < 8) ? 0 : 8);
    const int c  = K & 7;
    const int j  = K >> 3;                    // g*9 + k
    const int g  = j / 9;
    const int kk = j - g * 9;
    const int ci = g * CG + c;
    const int o  = coT * 16 + (l & 15);
    wpf[i] = (_Float16)w[((size_t)o * CCH + ci) * 9 + kk];
}

// forward-backward consistency check -> valid mask (1 channel)
__global__ __launch_bounds__(256)
void k_fbcheck(const float* __restrict__ fp, const float* __restrict__ fc,
               float* __restrict__ valid) {
    int p = blockIdx.x * blockDim.x + threadIdx.x;
    if (p >= HWP) return;
    const int y = p / WW, x = p - y * WW;
    const float fx = fp[p], fy = fp[HWP + p];
    const float xs = x + fx, ys = y + fy;
    const float bwx = bilin(fc, xs, ys);
    const float bwy = bilin(fc + HWP, xs, ys);
    const float dx = fx + bwx, dy = fy + bwy;
    const float lsq = dx * dx + dy * dy;
    const float thr = 0.01f * ((fx * fx + fy * fy) + (bwx * bwx + bwy * bwy)) + 0.5f;
    valid[p] = (lsq < thr) ? 1.f : 0.f;
}

// warp a C-channel planar f32 tensor by flow fp (2 channels)
__global__ __launch_bounds__(256)
void k_flow_warp(const float* __restrict__ src, const float* __restrict__ fp,
                 float* __restrict__ dst, int n /* = C*HW */) {
    int i = blockIdx.x * blockDim.x + threadIdx.x;
    if (i >= n) return;
    const int c = i / HWP, p = i - c * HWP;
    const int y = p / WW, x = p - y * WW;
    const float xs = x + fp[p], ys = y + fp[HWP + p];
    dst[i] = bilin(src + (size_t)c * HWP, xs, ys);
}

// offsets: dy/dx = 3*tanh(out4) + flow_yx ; modm = sigmoid(out4[288+j])
__global__ __launch_bounds__(256)
void k_offset_finalize(const float* __restrict__ o4, const float* __restrict__ fp,
                       float* __restrict__ dyb, float* __restrict__ dxb,
                       float* __restrict__ modb) {
    int i = blockIdx.x * blockDim.x + threadIdx.x;
    if (i >= 144 * HWP) return;
    const int j = i / HWP, p = i - j * HWP;
    dyb[i]  = 3.f * tanhf(o4[(size_t)(2 * j    ) * HWP + p]) + fp[HWP + p];
    dxb[i]  = 3.f * tanhf(o4[(size_t)(2 * j + 1) * HWP + p]) + fp[p];
    modb[i] = 1.f / (1.f + expf(-o4[(size_t)(288 + j) * HWP + p]));
}

// modulated deformable sampling -> S f16 HWC [HW][1152], ch = (g*9+k)*8+c
// one thread per (j = g*9+k, pixel): 8 channels share bilinear weights
__global__ __launch_bounds__(256)
void k_deform_sample(const float* __restrict__ feat, const float* __restrict__ dyb,
                     const float* __restrict__ dxb, const float* __restrict__ modb,
                     _Float16* __restrict__ S) {
    int i = blockIdx.x * blockDim.x + threadIdx.x;
    if (i >= 144 * HWP) return;
    const int j = i / HWP, p = i - j * HWP;
    const int g = j / 9, k = j - g * 9;
    const int y = p / WW, x = p - y * WW;
    const float ys = (float)(y + k / 3 - 1) + dyb[i];
    const float xs = (float)(x + k % 3 - 1) + dxb[i];
    const float mm = modb[i];

    const float x0f = floorf(xs), y0f = floorf(ys);
    const int   x0 = (int)x0f,   y0 = (int)y0f;
    const float wx1 = xs - x0f,  wy1 = ys - y0f;
    const float wx0 = 1.f - wx1, wy0 = 1.f - wy1;
    const bool vx0 = (x0 >= 0) & (x0 <= WW - 1);
    const bool vx1 = (x0 + 1 >= 0) & (x0 + 1 <= WW - 1);
    const bool vy0 = (y0 >= 0) & (y0 <= HH - 1);
    const bool vy1 = (y0 + 1 >= 0) & (y0 + 1 <= HH - 1);
    const float w00 = (vy0 & vx0) ? wy0 * wx0 : 0.f;
    const float w01 = (vy0 & vx1) ? wy0 * wx1 : 0.f;
    const float w10 = (vy1 & vx0) ? wy1 * wx0 : 0.f;
    const float w11 = (vy1 & vx1) ? wy1 * wx1 : 0.f;
    const int o00 = (vy0 & vx0) ? y0 * WW + x0           : 0;
    const int o01 = (vy0 & vx1) ? y0 * WW + x0 + 1       : 0;
    const int o10 = (vy1 & vx0) ? (y0 + 1) * WW + x0     : 0;
    const int o11 = (vy1 & vx1) ? (y0 + 1) * WW + x0 + 1 : 0;

    const float* base = feat + (size_t)g * CG * HWP;
    v8h outv;
#pragma unroll
    for (int c = 0; c < 8; ++c) {
        const float* fp2 = base + (size_t)c * HWP;
        const float v = fp2[o00] * w00 + fp2[o01] * w01
                      + fp2[o10] * w10 + fp2[o11] * w11;
        outv[c] = (_Float16)(v * mm);
    }
    *(v8h*)(S + (size_t)p * 1152 + j * 8) = outv;   // 16B packed store
}

// ---------------------------------------------------------------------------
extern "C" void kernel_launch(void* const* d_in, const int* in_sizes, int n_in,
                              void* d_out, int out_size, void* d_ws, size_t ws_size,
                              hipStream_t stream) {
    (void)in_sizes; (void)n_in; (void)out_size; (void)ws_size;
    const float* x    = (const float*)d_in[0];
    const float* ff   = (const float*)d_in[1];
    const float* fbw  = (const float*)d_in[2];
    const float* mask = (const float*)d_in[3];

    // params flattened as a jax pytree: keys sorted alphabetically
    struct Dir {
        const float *b_b1,*b_b2,*b_w1,*b_w2,*da_b,*da_w,
                    *off_b1,*off_b2,*off_b3,*off_b4,
                    *off_w1,*off_w2,*off_w3,*off_w4;
    };
    auto getDir = [&](int base) {
        Dir d;
        d.b_b1  = (const float*)d_in[base + 0];
        d.b_b2  = (const float*)d_in[base + 1];
        d.b_w1  = (const float*)d_in[base + 2];
        d.b_w2  = (const float*)d_in[base + 3];
        d.da_b  = (const float*)d_in[base + 4];
        d.da_w  = (const float*)d_in[base + 5];
        d.off_b1= (const float*)d_in[base + 6];
        d.off_b2= (const float*)d_in[base + 7];
        d.off_b3= (const float*)d_in[base + 8];
        d.off_b4= (const float*)d_in[base + 9];
        d.off_w1= (const float*)d_in[base +10];
        d.off_w2= (const float*)d_in[base +11];
        d.off_w3= (const float*)d_in[base +12];
        d.off_w4= (const float*)d_in[base +13];
        return d;
    };
    const Dir Pb = getDir(4);                       // 'bwd'
    const float* f_b1 = (const float*)d_in[18];
    const float* f_b2 = (const float*)d_in[19];
    const float* f_w1 = (const float*)d_in[20];
    const float* f_w2 = (const float*)d_in[21];
    const Dir Pf = getDir(22);                      // 'fwd'

    // channel paddings: Cip mult of 32, CoP mult of 64
    const int CP_COND = 288;   // 261/258 -> 288
    const int CP_128  = 128;
    const int CP_DA   = 1152;
    const int COP_128 = 128;
    const int COP_432 = 448;   // 432 -> 448

    // ---- workspace bump allocator ----
    char* ws = (char*)d_ws;
    size_t off = 0;
    auto alloc = [&](size_t bytes) -> void* {
        void* p = ws + off;
        off = (off + bytes + 255) & ~(size_t)255;
        return p;
    };
    struct DirPack { _Float16 *o1,*o2,*o3,*o4,*b1,*b2,*da; };
    auto allocPack = [&]() {
        DirPack k;
        k.o1 = (_Float16*)alloc((size_t)9 * CP_COND * COP_128 * 2);
        k.o2 = (_Float16*)alloc((size_t)9 * CP_128  * COP_128 * 2);
        k.o3 = (_Float16*)alloc((size_t)9 * CP_128  * COP_128 * 2);
        k.o4 = (_Float16*)alloc((size_t)9 * CP_128  * COP_432 * 2);
        k.b1 = (_Float16*)alloc((size_t)9 * CP_COND * COP_128 * 2);
        k.b2 = (_Float16*)alloc((size_t)9 * CP_128  * COP_128 * 2);
        k.da = (_Float16*)alloc((size_t)CP_DA * COP_128 * 2);
        return k;
    };
    DirPack Kb = allocPack(), Kf = allocPack();
    _Float16* Kf1 = (_Float16*)alloc((size_t)9 * CP_COND * COP_128 * 2);
    _Float16* Kf2 = (_Float16*)alloc((size_t)9 * CP_128  * COP_128 * 2);

    _Float16* fA = (_Float16*)alloc((size_t)HWP * CP_COND * 2); // concat input (HWC)
    _Float16* fB = (_Float16*)alloc((size_t)HWP * CP_128  * 2);
    _Float16* fS = (_Float16*)alloc((size_t)HWP * CP_DA   * 2); // deform samples
    float* warped = (float*)alloc((size_t)128 * HWP * 4);
    float* bufT0  = (float*)alloc((size_t)128 * HWP * 4);
    float* bufT1  = (float*)alloc((size_t)128 * HWP * 4);
    float* fpn    = (float*)alloc((size_t)128 * HWP * 4);
    float* out4   = (float*)alloc((size_t)432 * HWP * 4);
    float* dyb    = (float*)alloc((size_t)144 * HWP * 4);
    float* dxb    = (float*)alloc((size_t)144 * HWP * 4);
    float* modb   = (float*)alloc((size_t)144 * HWP * 4);
    float* validb = (float*)alloc((size_t)HWP * 4);

    // ---- launch helpers ----
    auto packW = [&](const float* w, _Float16* wp, int Ci, int Cip, int Co, int CoP) {
        int n = 9 * Cip * CoP;
        k_pack_frag<<<dim3((n + 255) / 256), 256, 0, stream>>>(w, wp, Ci, Cip, Co, CoP);
    };
    auto convGrid = [&](int CoP) {
        int total = NPIXT * (CoP >> 6);
        return dim3((total + WVB - 1) / WVB);
    };
    auto conv9 = [&](const _Float16* in, const _Float16* wp, const float* bias,
                     const float* resid, float* o, int Cip, int Co, int CoP,
                     int fuse, float slope) {
        k_conv_wmma<9><<<convGrid(CoP), 128, 0, stream>>>(in, wp, bias, resid, o,
                                                          Cip, Co, CoP, fuse, slope);
    };
    auto conv1 = [&](const _Float16* in, const _Float16* wp, const float* bias,
                     float* o, int Cip, int Co) {
        k_conv_wmma<1><<<convGrid(Co), 128, 0, stream>>>(in, wp, bias, nullptr, o,
                                                         Cip, Co, Co, 0, 0.f);
    };
    auto tof16 = [&](const float* src, _Float16* dstBase, int chanOff, int C, int Cip) {
        int n = C * HWP;
        k_f16hwc<<<dim3((n + 255) / 256), 256, 0, stream>>>(src, dstBase, chanOff, C, Cip);
    };

    // zero pad channels of the concat buffer once (pad weights are zero, but
    // keep pad activations finite/clean)
    {
        int n = HWP * CP_COND;
        k_zero16<<<dim3((n + 255) / 256), 256, 0, stream>>>(fA, n);
    }

    // ---- pack all weights (f32 -> f16, fragment layout) ----
    packW(Pb.off_w1, Kb.o1, 261, CP_COND, 128, COP_128);
    packW(Pb.off_w2, Kb.o2, 128, CP_128, 128, COP_128);
    packW(Pb.off_w3, Kb.o3, 128, CP_128, 128, COP_128);
    packW(Pb.off_w4, Kb.o4, 128, CP_128, 432, COP_432);
    packW(Pb.b_w1,   Kb.b1, 258, CP_COND, 128, COP_128);
    packW(Pb.b_w2,   Kb.b2, 128, CP_128, 128, COP_128);
    k_pack_daw_frag<<<dim3((1152 * 128 + 255) / 256), 256, 0, stream>>>(Pb.da_w, Kb.da);
    packW(Pf.off_w1, Kf.o1, 261, CP_COND, 128, COP_128);
    packW(Pf.off_w2, Kf.o2, 128, CP_128, 128, COP_128);
    packW(Pf.off_w3, Kf.o3, 128, CP_128, 128, COP_128);
    packW(Pf.off_w4, Kf.o4, 128, CP_128, 432, COP_432);
    packW(Pf.b_w1,   Kf.b1, 258, CP_COND, 128, COP_128);
    packW(Pf.b_w2,   Kf.b2, 128, CP_128, 128, COP_128);
    k_pack_daw_frag<<<dim3((1152 * 128 + 255) / 256), 256, 0, stream>>>(Pf.da_w, Kf.da);
    packW(f_w1, Kf1, 258, CP_COND, 128, COP_128);
    packW(f_w2, Kf2, 128, CP_128, 128, COP_128);

    // ---- output regions ----
    float* obBase  = (float*)d_out;
    float* ofBase  = obBase + (size_t)TT * CCH * HWP;
    float* finBase = obBase + (size_t)2 * TT * CCH * HWP;

    auto backbone = [&](const float* featCur, const float* featPropIn,
                        const float* maskCur, const Dir& P, const DirPack& KP,
                        float* outSlot) {
        tof16(featCur,    fA, 0,   128, CP_COND);
        tof16(featPropIn, fA, 128, 128, CP_COND);
        tof16(maskCur,    fA, 256, 2,   CP_COND);
        conv9(fA, KP.b1, P.b_b1, nullptr, bufT0, CP_COND, 128, COP_128, 1, 0.2f);
        tof16(bufT0, fB, 0, 128, CP_128);
        conv9(fB, KP.b2, P.b_b2, featPropIn, outSlot, CP_128, 128, COP_128, 2, 0.f);
    };

    auto runPass = [&](const Dir& P, const DirPack& KP, bool isFwd,
                       const float* featsSrc, float* outB) {
        const float* prev = nullptr;
        for (int s = 0; s < TT; ++s) {
            const int fi = isFwd ? s : (TT - 1 - s);
            const float* featCur = featsSrc + (size_t)fi * CCH * HWP;
            const float* maskCur = mask + (size_t)fi * 2 * HWP;
            float* outSlot = outB + (size_t)fi * CCH * HWP;
            if (s == 0) {
                backbone(featCur, featCur, maskCur, P, KP, outSlot);
                prev = outSlot;
                continue;
            }
            const float* fp = isFwd ? (fbw + (size_t)(s - 1) * 2 * HWP)
                                    : (ff  + (size_t)fi * 2 * HWP);
            const float* fc = isFwd ? (ff  + (size_t)(s - 1) * 2 * HWP)
                                    : (fbw + (size_t)fi * 2 * HWP);
            // fb_check -> valid, flow_warp of previous propagated features
            k_fbcheck<<<dim3((HWP + 255) / 256), 256, 0, stream>>>(fp, fc, validb);
            k_flow_warp<<<dim3((128 * HWP + 255) / 256), 256, 0, stream>>>(prev, fp, warped, 128 * HWP);
            // cond = cat(feat_cur, warped, flow, valid, mask) : 261 ch (HWC f16)
            tof16(featCur, fA, 0,   128, CP_COND);
            tof16(warped,  fA, 128, 128, CP_COND);
            tof16(fp,      fA, 256, 2,   CP_COND);
            tof16(validb,  fA, 258, 1,   CP_COND);
            tof16(maskCur, fA, 259, 2,   CP_COND);
            // offset conv stack
            conv9(fA, KP.o1, P.off_b1, nullptr, bufT0, CP_COND, 128, COP_128, 1, 0.1f);
            tof16(bufT0, fB, 0, 128, CP_128);
            conv9(fB, KP.o2, P.off_b2, nullptr, bufT1, CP_128, 128, COP_128, 1, 0.1f);
            tof16(bufT1, fB, 0, 128, CP_128);
            conv9(fB, KP.o3, P.off_b3, nullptr, bufT0, CP_128, 128, COP_128, 1, 0.1f);
            tof16(bufT0, fB, 0, 128, CP_128);
            conv9(fB, KP.o4, P.off_b4, nullptr, out4, CP_128, 432, COP_432, 0, 0.f);
            // offsets + modulation, deformable sampling, grouped einsum (WMMA)
            k_offset_finalize<<<dim3((144 * HWP + 255) / 256), 256, 0, stream>>>(out4, fp, dyb, dxb, modb);
            k_deform_sample<<<dim3((144 * HWP + 255) / 256), 256, 0, stream>>>(prev, dyb, dxb, modb, fS);
            conv1(fS, KP.da, P.da_b, fpn, CP_DA, 128);
            // residual backbone
            backbone(featCur, fpn, maskCur, P, KP, outSlot);
            prev = outSlot;
        }
    };

    // backward pass over x (reversed), forward pass over ob
    runPass(Pb, Kb, false, x, obBase);
    runPass(Pf, Kf, true, obBase, ofBase);

    // final fusion per frame: cat(ob, of, mask) -> conv -> conv + x
    for (int t = 0; t < TT; ++t) {
        const float* obT = obBase + (size_t)t * CCH * HWP;
        const float* ofT = ofBase + (size_t)t * CCH * HWP;
        const float* mT  = mask + (size_t)t * 2 * HWP;
        const float* xT  = x + (size_t)t * CCH * HWP;
        float* oT = finBase + (size_t)t * CCH * HWP;
        tof16(obT, fA, 0,   128, CP_COND);
        tof16(ofT, fA, 128, 128, CP_COND);
        tof16(mT,  fA, 256, 2,   CP_COND);
        conv9(fA, Kf1, f_b1, nullptr, bufT0, CP_COND, 128, COP_128, 1, 0.2f);
        tof16(bufT0, fB, 0, 128, CP_128);
        conv9(fB, Kf2, f_b2, xT, oT, CP_128, 128, COP_128, 2, 0.f);
    }
}
